// AttentionHead_26199300505858
// MI455X (gfx1250) — compile-verified, matrix operand order checked by reading, not run on hardware
//
#include <hip/hip_runtime.h>

// CDNA5 (gfx1250) wave32 WMMA fragment types
typedef __attribute__((ext_vector_type(16))) __bf16 v16bf;
typedef __attribute__((ext_vector_type(8)))  __bf16 v8bf;
typedef __attribute__((ext_vector_type(8)))  float  v8f;

#define B_   4
#define S_   2048
#define D_   1024
#define H_   16
#define HD_  64
#define N3_  (3 * D_)
#define M_   (B_ * S_)   // 8192 total rows

// Build a 16x32 bf16 WMMA fragment from two contiguous 16-byte runs.
// A-matrix layout (ISA 7.12.2): lane = 16*half + m; elems 0-7 -> K = 8*half+(0..7),
// elems 8-15 -> K = 16 + 8*half + (0..7).  B-matrix symmetric with lane -> N.
__device__ __forceinline__ v16bf mk_frag(const __bf16* p0, const __bf16* p1) {
    union { v16bf v; v8bf h[2]; } u;
    u.h[0] = *(const v8bf*)p0;
    u.h[1] = *(const v8bf*)p1;
    return u.v;
}

// For LDS generic pointers, addr[31:0] IS the LDS byte address (ISA 10.2 aperture
// table), so truncation gives the VDST operand
// for global_load_async_to_lds_*.
__device__ __forceinline__ unsigned lds_lo32(const void* p) {
    return (unsigned)(unsigned long long)p;
}

// Async copy 16 bytes/lane from global to LDS (tracked by ASYNCcnt).
__device__ __forceinline__ void async_b128(unsigned lds_addr, const __bf16* gp) {
    unsigned long long ga = (unsigned long long)gp;
    asm volatile("global_load_async_to_lds_b128 %0, %1, off"
                 :: "v"(lds_addr), "v"(ga) : "memory");
}

// ---------------- f32 -> bf16 converter ----------------
__global__ void k_cvt(const float* __restrict__ in, __bf16* __restrict__ out, int n) {
    int i = blockIdx.x * blockDim.x + threadIdx.x;
    int stride = gridDim.x * blockDim.x;
    for (; i < n; i += stride) out[i] = (__bf16)in[i];
}

// Tiled transpose+convert: out[c*rows + r] = bf16(in[r*cols + c]).
// 16x16 tiles via LDS; coalesced on both sides; no integer division.
__global__ __launch_bounds__(256)
void k_transpose_cvt(const float* __restrict__ in, __bf16* __restrict__ out,
                     int rows, int cols) {
    __shared__ float tile[16][17];
    const int tx = threadIdx.x & 15;
    const int ty = threadIdx.x >> 4;
    const int r0 = blockIdx.y * 16;
    const int c0 = blockIdx.x * 16;
    tile[ty][tx] = in[(size_t)(r0 + ty) * cols + (c0 + tx)];
    __syncthreads();
    out[(size_t)(c0 + ty) * rows + (r0 + tx)] = (__bf16)tile[tx][ty];
}

// ---------------- QKV projection GEMM: qkv = x @ w_in + b_in ----------------
// Each wave computes a 16x64 tile; epilogue scatters into Q [B,H,S,hd],
// K [B,H,S,hd], and V transposed [B,H,hd,S] (so PV fragments load contiguously).
__global__ __launch_bounds__(128)
void k_qkv_gemm(const __bf16* __restrict__ xb, const __bf16* __restrict__ wT,
                const float* __restrict__ bias,
                __bf16* __restrict__ qb, __bf16* __restrict__ kbuf,
                __bf16* __restrict__ vtb) {
    const int lane = threadIdx.x & 31;
    const int wave = threadIdx.x >> 5;
    const int hl   = lane >> 4;
    const int ln16 = lane & 15;

    const int m0 = (blockIdx.x * 4 + wave) * 16;
    const int n0 = blockIdx.y * 64;

    v8f acc[4] = {};
    for (int kk = 0; kk < D_; kk += 32) {
        const __bf16* arow = xb + (size_t)(m0 + ln16) * D_ + kk;
        __builtin_prefetch(arow + 256, 0, 3);   // unconditional: no branch in hot loop
        v16bf af = mk_frag(arow + hl * 8, arow + 16 + hl * 8);
        v16bf bf[4];
#pragma unroll
        for (int nt = 0; nt < 4; ++nt) {        // issue all B loads before any WMMA
            const __bf16* brow = wT + (size_t)(n0 + nt * 16 + ln16) * D_ + kk;
            bf[nt] = mk_frag(brow + hl * 8, brow + 16 + hl * 8);
        }
#pragma unroll
        for (int nt = 0; nt < 4; ++nt)
            acc[nt] = __builtin_amdgcn_wmma_f32_16x16x32_bf16(
                false, af, false, bf[nt], (short)0, acc[nt], false, false);
    }
    // epilogue: power-of-two / wave-uniform index math only (no div emulation)
    const int bidx  = m0 >> 11;          // m0 / S_
    const int sbase = m0 & (S_ - 1);
#pragma unroll
    for (int nt = 0; nt < 4; ++nt) {
        const int ncol  = n0 + nt * 16;      // wave-uniform, 16-aligned
        const int which = ncol >> 10;        // 0=q 1=k 2=v
        const int dd    = ncol & (D_ - 1);
        const int h     = dd >> 6;
        const int e     = (dd & 63) + ln16;  // hd index for this lane
        const float bv  = bias[ncol + ln16];
        const size_t bh = (size_t)(bidx * H_ + h);
#pragma unroll
        for (int r = 0; r < 8; ++r) {
            int s = sbase + r + 8 * hl;       // D-layout: vgpr r -> m = r + 8*half
            __bf16 o = (__bf16)(acc[nt][r] + bv);
            if (which == 0)      qb  [(bh * S_ + s) * HD_ + e] = o;
            else if (which == 1) kbuf[(bh * S_ + s) * HD_ + e] = o;
            else                 vtb [(bh * HD_ + e) * S_ + s] = o;
        }
    }
}

// ---------------- Flash attention (causal), one wave per 16-query tile ----------
// K/V tiles double-buffered in LDS via async-to-LDS engine (ASYNCcnt).
__global__ __launch_bounds__(32)
void k_attn(const __bf16* __restrict__ qb, const __bf16* __restrict__ kbuf,
            const __bf16* __restrict__ vtb, __bf16* __restrict__ attn) {
    __shared__ __bf16 Klds[2][32 * HD_];   // 32 keys x 64 hd, row-major (4 KB each)
    __shared__ __bf16 Vlds[2][HD_ * 32];   // 64 hd   x 32 keys, row-major (4 KB each)
    __shared__ __bf16 Plds[16 * 32];       // P tile: D-layout -> A-layout staging

    const int lane = threadIdx.x & 31;
    const int hl   = lane >> 4;
    const int ln16 = lane & 15;
    const int bh = blockIdx.x;             // b*H + h
    const int b  = bh >> 4;
    const int h  = bh & (H_ - 1);
    const int q0 = blockIdx.y * 16;

    const __bf16* qbase = qb   + (size_t)bh * S_ * HD_;
    const __bf16* kbase = kbuf + (size_t)bh * S_ * HD_;
    const __bf16* vbase = vtb  + (size_t)bh * HD_ * S_;

    // Q fragments for hd=64 (two K=32 steps), loaded once
    const __bf16* qrow = qbase + (size_t)(q0 + ln16) * HD_;
    v16bf qf0 = mk_frag(qrow + hl * 8,      qrow + 16 + hl * 8);
    v16bf qf1 = mk_frag(qrow + 32 + hl * 8, qrow + 48 + hl * 8);

    // async prefetch of one 32-key block (8 + 8 b128 instructions = 16 ASYNCcnt)
    auto prefetch = [&](int buf, int kb0) {
#pragma unroll
        for (int t = 0; t < 8; ++t) {             // K tile: 256 x 16B chunks
            int c   = t * 32 + lane;
            int row = c >> 3, ch = c & 7;         // 8 chunks per 128B row
            async_b128(lds_lo32(&Klds[buf][row * HD_ + ch * 8]),
                       kbase + (size_t)(kb0 + row) * HD_ + ch * 8);
        }
#pragma unroll
        for (int t = 0; t < 8; ++t) {             // V^T tile: 64 rows x 64B
            int c   = t * 32 + lane;
            int row = c >> 2, ch = c & 3;         // 4 chunks per 64B row
            async_b128(lds_lo32(&Vlds[buf][row * 32 + ch * 8]),
                       vbase + (size_t)row * S_ + kb0 + ch * 8);
        }
    };

    v8f o[4] = {};
    float mrow[8], lrow[8];
#pragma unroll
    for (int r = 0; r < 8; ++r) { mrow[r] = -3.0e38f; lrow[r] = 0.0f; }
    const float scale = 0.125f;   // 1/sqrt(64)

    const int nblocks = (q0 + 47) >> 5;   // covers keys 0 .. q0+15
    prefetch(0, 0);

    for (int ib = 0; ib < nblocks; ++ib) {
        const int kb0 = ib * 32;
        const int cur = ib & 1;
        if (ib + 1 < nblocks) {
            prefetch(cur ^ 1, kb0 + 32);
            // in-order retirement: <=16 outstanding => current tile resident
            asm volatile("s_wait_asynccnt 16" ::: "memory");
        } else {
            asm volatile("s_wait_asynccnt 0" ::: "memory");
        }

        // ---- scores: two 16-key subtiles, each = Q(16x64) x K^T(64x16), from LDS
        v8f sc[2];
#pragma unroll
        for (int t = 0; t < 2; ++t) {
            const __bf16* krow = &Klds[cur][(t * 16 + ln16) * HD_];
            v16bf kf0 = mk_frag(krow + hl * 8,      krow + 16 + hl * 8);
            v16bf kf1 = mk_frag(krow + 32 + hl * 8, krow + 48 + hl * 8);
            v8f s = {};
            s = __builtin_amdgcn_wmma_f32_16x16x32_bf16(false, qf0, false, kf0, (short)0, s, false, false);
            s = __builtin_amdgcn_wmma_f32_16x16x32_bf16(false, qf1, false, kf1, (short)0, s, false, false);
            sc[t] = s;
        }
        // ---- online softmax (row stats replicated across each 16-lane half)
#pragma unroll
        for (int r = 0; r < 8; ++r) {
            int qg = q0 + r + 8 * hl;
            float v0 = sc[0][r] * scale;
            float v1 = sc[1][r] * scale;
            if (kb0 + ln16      > qg) v0 = -3.0e38f;   // causal mask
            if (kb0 + 16 + ln16 > qg) v1 = -3.0e38f;
            float mx = fmaxf(v0, v1);
            mx = fmaxf(mx, __shfl_xor(mx, 1, 32));
            mx = fmaxf(mx, __shfl_xor(mx, 2, 32));
            mx = fmaxf(mx, __shfl_xor(mx, 4, 32));
            mx = fmaxf(mx, __shfl_xor(mx, 8, 32));
            float mnew = fmaxf(mrow[r], mx);
            float c  = __expf(mrow[r] - mnew);
            float p0 = __expf(v0 - mnew);
            float p1 = __expf(v1 - mnew);
            float sum = p0 + p1;
            sum += __shfl_xor(sum, 1, 32);
            sum += __shfl_xor(sum, 2, 32);
            sum += __shfl_xor(sum, 4, 32);
            sum += __shfl_xor(sum, 8, 32);
            lrow[r] = lrow[r] * c + sum;
            mrow[r] = mnew;
#pragma unroll
            for (int j = 0; j < 4; ++j) o[j][r] *= c;
            int m = r + 8 * hl;
            Plds[m * 32 + ln16]      = (__bf16)p0;
            Plds[m * 32 + 16 + ln16] = (__bf16)p1;
        }
        __syncthreads();
        // ---- PV: A = P (16x32, A-layout from LDS), B = V^T rows (LDS, contiguous)
        v16bf pf = mk_frag(&Plds[ln16 * 32 + hl * 8], &Plds[ln16 * 32 + 16 + hl * 8]);
#pragma unroll
        for (int j = 0; j < 4; ++j) {
            const __bf16* vrow = &Vlds[cur][(j * 16 + ln16) * 32];
            v16bf vf = mk_frag(vrow + hl * 8, vrow + 16 + hl * 8);
            o[j] = __builtin_amdgcn_wmma_f32_16x16x32_bf16(
                false, pf, false, vf, (short)0, o[j], false, false);
        }
        __syncthreads();
    }
    // ---- normalize and store to attn [B*S, D] (bf16, feeds output GEMM)
#pragma unroll
    for (int j = 0; j < 4; ++j) {
        int col = h * HD_ + j * 16 + ln16;
#pragma unroll
        for (int r = 0; r < 8; ++r) {
            int row = b * S_ + q0 + r + 8 * hl;
            attn[(size_t)row * D_ + col] = (__bf16)(o[j][r] / lrow[r]);
        }
    }
}

// ---------------- Output projection GEMM: out = attn @ w_out + b_out (f32 out) ----------------
__global__ __launch_bounds__(128)
void k_out_gemm(const __bf16* __restrict__ a, const __bf16* __restrict__ wT,
                const float* __restrict__ bias, float* __restrict__ out) {
    const int lane = threadIdx.x & 31;
    const int wave = threadIdx.x >> 5;
    const int hl   = lane >> 4;
    const int ln16 = lane & 15;
    const int m0 = (blockIdx.x * 4 + wave) * 16;
    const int n0 = blockIdx.y * 64;

    v8f acc[4] = {};
    for (int kk = 0; kk < D_; kk += 32) {
        const __bf16* arow = a + (size_t)(m0 + ln16) * D_ + kk;
        __builtin_prefetch(arow + 256, 0, 3);
        v16bf af = mk_frag(arow + hl * 8, arow + 16 + hl * 8);
        v16bf bf[4];
#pragma unroll
        for (int nt = 0; nt < 4; ++nt) {
            const __bf16* brow = wT + (size_t)(n0 + nt * 16 + ln16) * D_ + kk;
            bf[nt] = mk_frag(brow + hl * 8, brow + 16 + hl * 8);
        }
#pragma unroll
        for (int nt = 0; nt < 4; ++nt)
            acc[nt] = __builtin_amdgcn_wmma_f32_16x16x32_bf16(
                false, af, false, bf[nt], (short)0, acc[nt], false, false);
    }
#pragma unroll
    for (int nt = 0; nt < 4; ++nt) {
        int ng = n0 + nt * 16 + ln16;
        float bv = bias[ng];
#pragma unroll
        for (int r = 0; r < 8; ++r) {
            int mg = m0 + r + 8 * hl;
            out[(size_t)mg * D_ + ng] = acc[nt][r] + bv;
        }
    }
}

extern "C" void kernel_launch(void* const* d_in, const int* in_sizes, int n_in,
                              void* d_out, int out_size, void* d_ws, size_t ws_size,
                              hipStream_t stream) {
    const float* x     = (const float*)d_in[0];
    const float* w_in  = (const float*)d_in[1];
    const float* b_in  = (const float*)d_in[2];
    const float* w_out = (const float*)d_in[3];
    const float* b_out = (const float*)d_in[4];
    float* out = (float*)d_out;

    // workspace layout (bf16 buffers), total ~90 MB
    char* ws = (char*)d_ws;
    size_t off = 0;
    __bf16* xb    = (__bf16*)(ws + off); off += (size_t)M_  * D_ * 2;   // x bf16
    __bf16* winT  = (__bf16*)(ws + off); off += (size_t)N3_ * D_ * 2;   // w_in^T  [3D, D]
    __bf16* woutT = (__bf16*)(ws + off); off += (size_t)D_  * D_ * 2;   // w_out^T [D, D]
    __bf16* qb    = (__bf16*)(ws + off); off += (size_t)M_  * D_ * 2;   // Q [B,H,S,hd]
    __bf16* kb    = (__bf16*)(ws + off); off += (size_t)M_  * D_ * 2;   // K [B,H,S,hd]
    __bf16* vtb   = (__bf16*)(ws + off); off += (size_t)M_  * D_ * 2;   // V [B,H,hd,S]
    __bf16* attn  = (__bf16*)(ws + off);                                 // attn out [B*S, D]

    k_cvt<<<4096, 256, 0, stream>>>(x, xb, M_ * D_);
    { dim3 g(N3_ / 16, D_ / 16);  k_transpose_cvt<<<g, 256, 0, stream>>>(w_in,  winT,  D_, N3_); }
    { dim3 g(D_ / 16,  D_ / 16);  k_transpose_cvt<<<g, 256, 0, stream>>>(w_out, woutT, D_, D_); }

    dim3 g1(M_ / 64, N3_ / 64);           // 128 x 48, 4 waves/block
    k_qkv_gemm<<<g1, 128, 0, stream>>>(xb, winT, b_in, qb, kb, vtb);

    dim3 ga(B_ * H_, S_ / 16);            // 64 x 128, 1 wave/block
    k_attn<<<ga, 32, 0, stream>>>(qb, kb, vtb, attn);

    dim3 g2(M_ / 64, D_ / 64);            // 128 x 16
    k_out_gemm<<<g2, 128, 0, stream>>>(attn, woutT, b_out, out);
}
